// MultiHeadAttention_72911364817505
// MI455X (gfx1250) — matrix operand order repaired and checked by reading, not compiled, not run on hardware
//
#include <hip/hip_runtime.h>

#define DM   512
#define DK   64
#define NH   8
#define BATCH 4
#define SLEN 2048
#define NTOK (BATCH * SLEN)   // 8192

typedef __attribute__((ext_vector_type(16))) __bf16        v16bf;
typedef __attribute__((ext_vector_type(8)))  float         v8f;
typedef __attribute__((ext_vector_type(16))) unsigned short u16x16;
typedef __attribute__((ext_vector_type(8)))  unsigned short u16x8;

__device__ __forceinline__ unsigned short f2bf(float f) {
    unsigned int u = __float_as_uint(f);
    u += 0x7FFFu + ((u >> 16) & 1u);      // round-to-nearest-even
    return (unsigned short)(u >> 16);
}

__device__ __forceinline__ v8f wmma_bf16(v16bf a, v16bf b, v8f c) {
    // v_wmma_f32_16x16x32_bf16: (neg_a, A, neg_b, B, c_mod, C, reuse_a, reuse_b)
    return __builtin_amdgcn_wmma_f32_16x16x32_bf16(false, a, false, b, (short)0, c,
                                                   false, false);
}

__device__ __forceinline__ v16bf cat8(u16x8 lo, u16x8 hi) {
    u16x16 t = __builtin_shufflevector(lo, hi, 0, 1, 2, 3, 4, 5, 6, 7,
                                       8, 9, 10, 11, 12, 13, 14, 15);
    return __builtin_bit_cast(v16bf, t);
}

__device__ __forceinline__ v8f vzero8() {
    v8f z;
#pragma unroll
    for (int i = 0; i < 8; ++i) z[i] = 0.0f;
    return z;
}

// ---------------------------------------------------------------------------
// Weight transpose + f32 -> bf16:  Wt[c][k] = bf16(W[k][c])
// ---------------------------------------------------------------------------
__global__ __launch_bounds__(256) void transpose_w(const float* __restrict__ W,
                                                   unsigned short* __restrict__ Wt) {
    int idx = blockIdx.x * 256 + threadIdx.x;   // 512*512 elements
    int k = idx >> 9;
    int c = idx & 511;
    Wt[(size_t)c * DM + k] = f2bf(W[idx]);
}

// ---------------------------------------------------------------------------
// Projection GEMM: P = X[8192,512](f32) @ W + b, emitted as bf16 in
// head-split layout.  mode 0: q -> [B,H,S,DK] * dk^-0.5
//                     mode 1: k -> [B,H,S,DK]
//                     mode 2: v -> [B,H,DK,S]  (transposed for P*V B-frags)
// One wave computes a 16x64 output tile (4 independent WMMA accumulators).
// ---------------------------------------------------------------------------
__global__ __launch_bounds__(256) void proj_gemm(const float* __restrict__ X,
                                                 const unsigned short* __restrict__ Wt,
                                                 const float* __restrict__ bias,
                                                 unsigned short* __restrict__ dst,
                                                 int mode) {
    const int widx = blockIdx.x * 8 + (threadIdx.x >> 5);  // 0..4095
    const int lane = threadIdx.x & 31;
    const int nl   = lane & 15;
    const int hl   = lane >> 4;
    const int row0 = (widx >> 3) * 16;   // 512 row tiles
    const int col0 = (widx & 7) * 64;    // 8 col groups of 64

    v8f acc[4];
#pragma unroll
    for (int j = 0; j < 4; ++j) acc[j] = vzero8();

    const float* xrow = X + (size_t)(row0 + nl) * DM;

    for (int kc = 0; kc < DM; kc += 32) {
        // A fragment: lane holds row (row0+nl); halves 0..15 = K kc+hl*16..
        // (contiguous, convert f32 -> bf16 in registers)
        const float* p = xrow + kc + hl * 16;
        float4 x0 = *(const float4*)(p);
        float4 x1 = *(const float4*)(p + 4);
        float4 x2 = *(const float4*)(p + 8);
        float4 x3 = *(const float4*)(p + 12);
        u16x16 ta;
        ta[0]  = f2bf(x0.x); ta[1]  = f2bf(x0.y); ta[2]  = f2bf(x0.z); ta[3]  = f2bf(x0.w);
        ta[4]  = f2bf(x1.x); ta[5]  = f2bf(x1.y); ta[6]  = f2bf(x1.z); ta[7]  = f2bf(x1.w);
        ta[8]  = f2bf(x2.x); ta[9]  = f2bf(x2.y); ta[10] = f2bf(x2.z); ta[11] = f2bf(x2.w);
        ta[12] = f2bf(x3.x); ta[13] = f2bf(x3.y); ta[14] = f2bf(x3.z); ta[15] = f2bf(x3.w);
        v16bf a = __builtin_bit_cast(v16bf, ta);

#pragma unroll
        for (int j = 0; j < 4; ++j) {
            // B fragment: lane = out col, 16 contiguous K halves (+16 upper half)
            const unsigned short* wp =
                Wt + (size_t)(col0 + j * 16 + nl) * DM + kc + hl * 16;
            v16bf b = __builtin_bit_cast(v16bf, *(const u16x16*)wp);
            acc[j] = wmma_bf16(a, b, acc[j]);
        }
    }

#pragma unroll
    for (int j = 0; j < 4; ++j) {
        const int c   = col0 + j * 16 + nl;
        const int h   = c >> 6;
        const int d   = c & 63;
        const float bc = bias[c];
#pragma unroll
        for (int r = 0; r < 8; ++r) {
            const int M = r + 8 * hl;
            const int t = row0 + M;
            const int bq = t >> 11;        // / SLEN
            const int s  = t & (SLEN - 1);
            float val = acc[j][r] + bc;
            if (mode == 0) {
                dst[((size_t)(bq * NH + h) * SLEN + s) * DK + d] = f2bf(val * 0.125f);
            } else if (mode == 1) {
                dst[((size_t)(bq * NH + h) * SLEN + s) * DK + d] = f2bf(val);
            } else {
                dst[((size_t)(bq * NH + h) * DK + d) * SLEN + s] = f2bf(val);
            }
        }
    }
}

// ---------------------------------------------------------------------------
// Flash attention: one wave owns a 16-row q-tile of one (b,h); streams keys in
// blocks of 32 with online softmax.  Per block: 4 *independent* score WMMAs
// (partials combined with packed f32 adds — avoids the 5-slot WMMA->WMMA RAW
// hazard) + 4 P*V WMMAs into independent O accumulators.  Row stats (m,l)
// live replicated across each 16-lane half, matching the 16x16 f32 C-fragment
// layout (row M = r + 8*(lane/16)).
// ---------------------------------------------------------------------------
__global__ __launch_bounds__(256) void mha_flash(const unsigned short* __restrict__ qb,
                                                 const unsigned short* __restrict__ kbuf,
                                                 const unsigned short* __restrict__ vt,
                                                 unsigned short* __restrict__ attn_out) {
    __shared__ unsigned short lds[8][16][32];  // per-wave 16x32 bf16 prob tile

    const int lane = threadIdx.x & 31;
    const int wave = threadIdx.x >> 5;
    const int nl   = lane & 15;
    const int hl   = lane >> 4;

    const int bh = blockIdx.x >> 4;        // 0..31  (b*8 + h)
    const int q0 = (blockIdx.x & 15) * 128 + wave * 16;
    const int bq = bh >> 3;
    const int h  = bh & 7;

    const unsigned short* qhead = qb   + (size_t)bh * SLEN * DK;
    const unsigned short* khead = kbuf + (size_t)bh * SLEN * DK;
    const unsigned short* vhead = vt   + (size_t)bh * DK * SLEN;

    // Q A-fragments for d = 0..31 and 32..63 (loaded once, reused 64x)
    v16bf qA[2];
#pragma unroll
    for (int dc2 = 0; dc2 < 2; ++dc2) {
        const unsigned short* p = qhead + (size_t)(q0 + nl) * DK + dc2 * 32 + hl * 16;
        u16x8 lo = *(const u16x8*)(p);
        u16x8 hi = *(const u16x8*)(p + 8);
        qA[dc2] = cat8(lo, hi);
    }

    float m[8], l[8];
    v8f   O[4];
#pragma unroll
    for (int r = 0; r < 8; ++r) { m[r] = -1e30f; l[r] = 0.0f; }
#pragma unroll
    for (int g = 0; g < 4; ++g) O[g] = vzero8();

    for (int kb0 = 0; kb0 < SLEN; kb0 += 32) {
        // prefetch next key block's lines (+4KB stride jump); lowers to
        // global_prefetch_b8, no LOADcnt cost, speculative-safe
        __builtin_prefetch(khead + (size_t)(kb0 + 32 + nl) * DK + hl * 16, 0, 3);
        __builtin_prefetch(khead + (size_t)(kb0 + 48 + nl) * DK + hl * 16, 0, 3);

        // ---- scores: 4 independent WMMAs, partials joined with v_pk_add_f32 ----
        v8f s[2];
#pragma unroll
        for (int ng = 0; ng < 2; ++ng) {
            v8f part[2];
#pragma unroll
            for (int dc2 = 0; dc2 < 2; ++dc2) {
                const unsigned short* p =
                    khead + (size_t)(kb0 + ng * 16 + nl) * DK + dc2 * 32 + hl * 16;
                v16bf b = __builtin_bit_cast(v16bf, *(const u16x16*)p);
                part[dc2] = wmma_bf16(qA[dc2], b, vzero8());
            }
            s[ng] = part[0] + part[1];
        }

        // ---- online softmax over the 32-key block ----
        float p0[8], p1[8];
#pragma unroll
        for (int r = 0; r < 8; ++r) {
            float mb = fmaxf(s[0][r], s[1][r]);
            mb = fmaxf(mb, __shfl_xor(mb, 1, 32));
            mb = fmaxf(mb, __shfl_xor(mb, 2, 32));
            mb = fmaxf(mb, __shfl_xor(mb, 4, 32));
            mb = fmaxf(mb, __shfl_xor(mb, 8, 32));
            const float mnew  = fmaxf(m[r], mb);
            const float alpha = __expf(m[r] - mnew);
            const float a0 = __expf(s[0][r] - mnew);
            const float a1 = __expf(s[1][r] - mnew);
            float rs = a0 + a1;
            rs += __shfl_xor(rs, 1, 32);
            rs += __shfl_xor(rs, 2, 32);
            rs += __shfl_xor(rs, 4, 32);
            rs += __shfl_xor(rs, 8, 32);
            l[r] = l[r] * alpha + rs;
            m[r] = mnew;
            p0[r] = a0;
            p1[r] = a1;
#pragma unroll
            for (int g = 0; g < 4; ++g) O[g][r] *= alpha;
        }

        // ---- C-layout probs -> A-fragment via wave-private LDS tile ----
#pragma unroll
        for (int r = 0; r < 8; ++r) {
            const int M = r + 8 * hl;
            lds[wave][M][nl]      = f2bf(p0[r]);
            lds[wave][M][nl + 16] = f2bf(p1[r]);
        }
        __builtin_amdgcn_wave_barrier();
        u16x8 plo = *(const u16x8*)&lds[wave][nl][hl * 16];
        u16x8 phi = *(const u16x8*)&lds[wave][nl][hl * 16 + 8];
        __builtin_amdgcn_wave_barrier();
        v16bf pA = cat8(plo, phi);

        // ---- O += P @ V  (V stored [DK,S]: contiguous key-halves per lane) ----
#pragma unroll
        for (int g = 0; g < 4; ++g) {
            const unsigned short* p =
                vhead + (size_t)(g * 16 + nl) * SLEN + kb0 + hl * 16;
            v16bf b = __builtin_bit_cast(v16bf, *(const u16x16*)p);
            O[g] = wmma_bf16(pA, b, O[g]);
        }
    }

    // ---- normalize and write bf16 attention output [NTOK, DM] ----
#pragma unroll
    for (int r = 0; r < 8; ++r) {
        const float inv = 1.0f / l[r];
        const int M = r + 8 * hl;
        const int t = bq * SLEN + q0 + M;
#pragma unroll
        for (int g = 0; g < 4; ++g) {
            const int c = h * DK + g * 16 + nl;
            attn_out[(size_t)t * DM + c] = f2bf(O[g][r] * inv);
        }
    }
}

// ---------------------------------------------------------------------------
// Output projection: out = A[8192,512](bf16) @ w_fc + b_fc  -> f32 d_out
// ---------------------------------------------------------------------------
__global__ __launch_bounds__(256) void out_gemm(const unsigned short* __restrict__ A,
                                                const unsigned short* __restrict__ Wt,
                                                const float* __restrict__ bias,
                                                float* __restrict__ out) {
    const int widx = blockIdx.x * 8 + (threadIdx.x >> 5);
    const int lane = threadIdx.x & 31;
    const int nl   = lane & 15;
    const int hl   = lane >> 4;
    const int row0 = (widx >> 3) * 16;
    const int col0 = (widx & 7) * 64;

    v8f acc[4];
#pragma unroll
    for (int j = 0; j < 4; ++j) acc[j] = vzero8();

    const unsigned short* arow = A + (size_t)(row0 + nl) * DM;

    for (int kc = 0; kc < DM; kc += 32) {
        const unsigned short* p = arow + kc + hl * 16;
        u16x8 lo = *(const u16x8*)(p);
        u16x8 hi = *(const u16x8*)(p + 8);
        v16bf a  = cat8(lo, hi);
#pragma unroll
        for (int j = 0; j < 4; ++j) {
            const unsigned short* wp =
                Wt + (size_t)(col0 + j * 16 + nl) * DM + kc + hl * 16;
            v16bf b = __builtin_bit_cast(v16bf, *(const u16x16*)wp);
            acc[j] = wmma_bf16(a, b, acc[j]);
        }
    }

#pragma unroll
    for (int j = 0; j < 4; ++j) {
        const int c = col0 + j * 16 + nl;
        const float bc = bias[c];
#pragma unroll
        for (int r = 0; r < 8; ++r) {
            const int t = row0 + r + 8 * hl;
            out[(size_t)t * DM + c] = acc[j][r] + bc;
        }
    }
}

// ---------------------------------------------------------------------------
extern "C" void kernel_launch(void* const* d_in, const int* in_sizes, int n_in,
                              void* d_out, int out_size, void* d_ws, size_t ws_size,
                              hipStream_t stream) {
    const float* Q    = (const float*)d_in[0];
    const float* K    = (const float*)d_in[1];
    const float* V    = (const float*)d_in[2];
    const float* w_q  = (const float*)d_in[3];
    const float* b_q  = (const float*)d_in[4];
    const float* w_k  = (const float*)d_in[5];
    const float* b_k  = (const float*)d_in[6];
    const float* w_v  = (const float*)d_in[7];
    const float* b_v  = (const float*)d_in[8];
    const float* w_fc = (const float*)d_in[9];
    const float* b_fc = (const float*)d_in[10];

    char* ws = (char*)d_ws;
    const size_t WSZ = (size_t)DM * DM * 2;            // 512 KB per bf16 weight
    unsigned short* wt_q  = (unsigned short*)(ws + 0 * WSZ);
    unsigned short* wt_k  = (unsigned short*)(ws + 1 * WSZ);
    unsigned short* wt_v  = (unsigned short*)(ws + 2 * WSZ);
    unsigned short* wt_fc = (unsigned short*)(ws + 3 * WSZ);
    const size_t ACT = (size_t)NTOK * DM * 2;          // 8 MB per bf16 activation
    unsigned short* qb   = (unsigned short*)(ws + 4 * WSZ);
    unsigned short* kbuf = (unsigned short*)(ws + 4 * WSZ + 1 * ACT);
    unsigned short* vt   = (unsigned short*)(ws + 4 * WSZ + 2 * ACT);
    unsigned short* attn = (unsigned short*)(ws + 4 * WSZ + 3 * ACT);

    transpose_w<<<(DM * DM) / 256, 256, 0, stream>>>(w_q, wt_q);
    transpose_w<<<(DM * DM) / 256, 256, 0, stream>>>(w_k, wt_k);
    transpose_w<<<(DM * DM) / 256, 256, 0, stream>>>(w_v, wt_v);
    transpose_w<<<(DM * DM) / 256, 256, 0, stream>>>(w_fc, wt_fc);

    // 512 row-tiles * 8 col-groups = 4096 waves = 512 blocks of 8 waves
    proj_gemm<<<512, 256, 0, stream>>>(Q, wt_q, b_q, qb, 0);
    proj_gemm<<<512, 256, 0, stream>>>(K, wt_k, b_k, kbuf, 1);
    proj_gemm<<<512, 256, 0, stream>>>(V, wt_v, b_v, vt, 2);

    // 32 (b,h) * 16 q-chunks of 128 rows
    mha_flash<<<512, 256, 0, stream>>>(qb, kbuf, vt, attn);

    out_gemm<<<512, 256, 0, stream>>>(attn, wt_fc, b_fc, (float*)d_out);
}
